// SharedIndependentNodeNetwork_72310069395693
// MI455X (gfx1250) — compile-verified, matrix operand order checked by reading, not run on hardware
//
#include <hip/hip_runtime.h>

typedef __attribute__((ext_vector_type(16))) _Float16 v16h;
typedef __attribute__((ext_vector_type(8)))  float    v8f;
typedef _Float16 half_t;

// Problem constants (from reference): B=8192, D=128, N=64, C=256
constexpr int Bc = 8192;
constexpr int Dc = 128;
constexpr int Nc = 64;
constexpr int Cc = 256;

constexpr int BPB     = 16;          // b's processed per block (amortize weight staging)
constexpr int NBLK    = Bc / BPB;    // 512 blocks
constexpr int THREADS = 256;         // 8 wave32 waves

// LDS strides (elements), padded against bank conflicts.
constexpr int W1_S = 132;            // w1t[c][d]   f16 : 256 x 132
constexpr int W2_S = 260;            // w2t[co][ci] f16 : 256 x 260
constexpr int H_S  = 260;            // h[n][c]     f16 : 64 x 260  (H1/H2 aliased)
constexpr int XF_S = 68;             // xf[d][n]    f32 : 128 x 68 (rows 272B = 16B-aligned)

struct __align__(16) Smem {
    half_t w1t[Cc * W1_S];           //  67,584 B
    half_t w2t[Cc * W2_S];           // 133,120 B
    float  xf [2][Dc * XF_S];        //  69,632 B (double-buffered async-DMA target)
    half_t h  [Nc * H_S];            //  33,280 B (layer-1 out, then layer-2 out)
    float  sb1[Cc];                  //   1,024 B
    float  sb2[Cc];                  //   1,024 B
    float  sw3[Cc];                  //   1,024 B
};                                   // ~300 KB of the WGP's 320 KB LDS

// 16-bit fragment gather from an LDS f16 tile with contiguous k within a row.
// Per ISA 7.12.2: lane (lr, hf), element v holds k = k0 + (v/4)*16 + hf*8 + (v&3)*2 (+e).
__device__ __forceinline__ v16h load_frag(const half_t* __restrict__ tile, int stride,
                                          int rc_base, int k0, int lr, int hf) {
    union { v16h v; unsigned u[8]; } f;
    const half_t* p = tile + (rc_base + lr) * stride + k0 + hf * 8;
#pragma unroll
    for (int v = 0; v < 8; ++v) {
        f.u[v] = *(const unsigned*)(p + ((v >> 2) * 16 + (v & 3) * 2));
    }
    return f.v;
}

// A-fragment built from the f32 X tile stored d-major: element (row n, k=d) lives at
// xf[k*XF_S + n]. Transpose + f32->f16 conversion happen here (once per wave per b).
__device__ __forceinline__ v16h load_frag_xf(const float* __restrict__ tile,
                                             int n, int k0, int hf) {
    union { v16h v; unsigned u[8]; } f;
    const float* p = tile + n + (k0 + hf * 8) * XF_S;
#pragma unroll
    for (int v = 0; v < 8; ++v) {
        int krel = ((v >> 2) * 16 + (v & 3) * 2) * XF_S;
        union { half_t h[2]; unsigned u; } pk;
        pk.h[0] = (half_t)p[krel];
        pk.h[1] = (half_t)p[krel + XF_S];
        f.u[v] = pk.u;
    }
    return f.v;
}

__global__ __launch_bounds__(THREADS, 1)
void mlp3_fused_kernel(const float* __restrict__ X,  const float* __restrict__ W1,
                       const float* __restrict__ b1, const float* __restrict__ W2,
                       const float* __restrict__ b2, const float* __restrict__ W3,
                       const float* __restrict__ b3, float* __restrict__ out) {
    __shared__ Smem sm;

    const int t    = threadIdx.x;
    const int lane = t & 31;
    const int wv   = t >> 5;         // 0..7
    const int lr   = lane & 15;
    const int hf   = lane >> 4;
    const int rt   = wv & 3;         // row tile: rows [16*rt, 16*rt+16)
    const int ch   = wv >> 2;        // column half: cols [128*ch, 128*ch+128)
    const int m0   = rt * 16;

    // Async-DMA one X batch tile (8192 f32 = 2048 16B chunks) into xf[buf].
    // Each thread issues 8 global_load_async_to_lds_b128 (ASYNCcnt-tracked, no VGPR data).
    auto issue_stage = [&](int bb, int buf) {
        const float* Xb = X + (size_t)(blockIdx.x * BPB + bb) * (Dc * Nc);
        unsigned lbase = (unsigned)(uintptr_t)(&sm.xf[buf][0]);
#pragma unroll
        for (int i = 0; i < (Dc * Nc) / (4 * THREADS); ++i) {
            int chunk = i * THREADS + t;          // 16B chunk index
            int d  = chunk >> 4;                  // 16 chunks per 64-float row
            int c4 = (chunk & 15) << 2;           // n offset within row
            unsigned lds  = lbase + (unsigned)((d * XF_S + c4) * 4);
            unsigned goff = (unsigned)(chunk << 4);   // bytes (rows contiguous in global)
            asm volatile("global_load_async_to_lds_b128 %0, %1, %2"
                         :: "v"(lds), "v"(goff), "s"(Xb) : "memory");
        }
    };

    // ---------------- stage weights/biases into LDS (once per block) ----------------
#pragma unroll 4
    for (int i = t; i < Dc * Cc; i += THREADS) {       // W1 (d-major) -> w1t[c][d]
        int d = i >> 8, c = i & 255;
        sm.w1t[c * W1_S + d] = (half_t)W1[i];
    }
#pragma unroll 4
    for (int i = t; i < Cc * Cc; i += THREADS) {       // W2 (ci-major) -> w2t[co][ci]
        int ci = i >> 8, co = i & 255;
        sm.w2t[co * W2_S + ci] = (half_t)W2[i];
    }
    sm.sb1[t] = b1[t];
    sm.sb2[t] = b2[t];
    sm.sw3[t] = W3[t];
    const float bias3 = b3[0];

    issue_stage(0, 0);                 // kick off first X tile while weights settle
    __syncthreads();

    for (int bb = 0; bb < BPB; ++bb) {
        const int b   = blockIdx.x * BPB + bb;
        const int cur = bb & 1;

        // Wait for this batch's DMA (per-wave counter), then make it block-visible.
        asm volatile("s_wait_asynccnt 0x0" ::: "memory");
        __syncthreads();

        // Fire off the next batch's DMA into the other buffer; it overlaps all
        // of this batch's compute below.
        if (bb + 1 < BPB) issue_stage(bb + 1, cur ^ 1);

        // ---------------- layer 1: H1 = relu(X^T @ W1 + b1), K = 128 ----------------
        v16h a1[4];
#pragma unroll
        for (int k = 0; k < 4; ++k) a1[k] = load_frag_xf(sm.xf[cur], m0 + lr, k * 32, hf);

#pragma unroll
        for (int ct = 0; ct < 8; ++ct) {
            const int c0 = ch * 128 + ct * 16;
            v8f acc = {};
#pragma unroll
            for (int k = 0; k < 4; ++k) {
                v16h bf = load_frag(sm.w1t, W1_S, c0, k * 32, lr, hf);
                acc = __builtin_amdgcn_wmma_f32_16x16x32_f16(
                          false, a1[k], false, bf, (short)0, acc, false, false);
            }
            const int col  = c0 + lr;
            const float bs = sm.sb1[col];
#pragma unroll
            for (int v = 0; v < 8; ++v) {
                float x = acc[v] + bs;
                x = x > 0.f ? x : 0.f;
                sm.h[(m0 + hf * 8 + v) * H_S + col] = (half_t)x;
            }
        }
        __syncthreads();

        // ---------------- layer 2: H2 = relu(H1 @ W2 + b2), K = 256 ----------------
        // H1 and H2 share sm.h: hoist ALL A-fragments, barrier, then overwrite.
        v16h a2[8];
#pragma unroll
        for (int k = 0; k < 8; ++k) a2[k] = load_frag(sm.h, H_S, m0, k * 32, lr, hf);
        __syncthreads();

#pragma unroll
        for (int ct = 0; ct < 8; ++ct) {
            const int c0 = ch * 128 + ct * 16;
            v8f acc = {};
#pragma unroll
            for (int k = 0; k < 8; ++k) {
                v16h bf = load_frag(sm.w2t, W2_S, c0, k * 32, lr, hf);
                acc = __builtin_amdgcn_wmma_f32_16x16x32_f16(
                          false, a2[k], false, bf, (short)0, acc, false, false);
            }
            const int col  = c0 + lr;
            const float bs = sm.sb2[col];
#pragma unroll
            for (int v = 0; v < 8; ++v) {
                float x = acc[v] + bs;
                x = x > 0.f ? x : 0.f;
                sm.h[(m0 + hf * 8 + v) * H_S + col] = (half_t)x;
            }
        }
        __syncthreads();

        // -------- layer 3: logits[n] = H2[n,:] . W3 + b3 ; out[b,0] = 0 --------
        if (t < Nc) {
            const half_t* __restrict__ hrow = sm.h + t * H_S;
            float s = bias3;
#pragma unroll 8
            for (int c = 0; c < Cc; ++c) s += (float)hrow[c] * sm.sw3[c];
            out[(size_t)b * (Nc + 1) + 1 + t] = s;
        } else if (t == Nc) {
            out[(size_t)b * (Nc + 1)] = 0.f;
        }
        __syncthreads();   // h consumed before next iteration overwrites it
    }
}

extern "C" void kernel_launch(void* const* d_in, const int* in_sizes, int n_in,
                              void* d_out, int out_size, void* d_ws, size_t ws_size,
                              hipStream_t stream) {
    (void)in_sizes; (void)n_in; (void)out_size; (void)d_ws; (void)ws_size;
    const float* X  = (const float*)d_in[0];
    const float* W1 = (const float*)d_in[1];
    const float* b1 = (const float*)d_in[2];
    const float* W2 = (const float*)d_in[3];
    const float* b2 = (const float*)d_in[4];
    const float* W3 = (const float*)d_in[5];
    const float* b3 = (const float*)d_in[6];
    float* out = (float*)d_out;

    mlp3_fused_kernel<<<NBLK, THREADS, 0, stream>>>(X, W1, b1, W2, b2, W3, b3, out);
}
